// BigBirdAttention_24799141167759
// MI455X (gfx1250) — matrix operand order repaired and checked by reading, not compiled, not run on hardware
//
#include <hip/hip_runtime.h>
#include <hip/hip_bf16.h>
#include <stdint.h>

// ---------------------------------------------------------------------------
// BigBird attention, MI455X (gfx1250), wave32 + WMMA bf16 path.
//   B=1, S=4096, C=512, H=8, Dh=64
// Pipeline:
//   1) f32 -> bf16 converts (x, Wq, Wk, Wv, Wo)
//   2) 64x64 block-occupancy reduction of the BigBird mask
//   3) WMMA GEMM: Q/K/V = x @ W^T + b          (bf16 out, [S, C] head-packed)
//   4) WMMA flash attention, skipping all-masked 64x64 KV blocks
//   5) WMMA GEMM: out = AttO @ Wo^T + bo        (f32 out)
// ---------------------------------------------------------------------------

typedef __attribute__((ext_vector_type(16))) __bf16 v16bf;
typedef __attribute__((ext_vector_type(8)))  float  v8f;

#define S_LEN 4096
#define C_DIM 512
#define NH    8
#define DH    64
#define NBLK  (S_LEN / 64)     // 64 KV/Q blocks

union Frag {                 // one WMMA 16x32 bf16 operand (8 VGPRs per lane)
  v16bf v;
  uint4 q[2];
  unsigned short u[16];
};

__device__ inline unsigned short f2bf(float f) {   // RNE float -> bf16
  unsigned int u = __float_as_uint(f);
  unsigned int r = u + 0x7FFFu + ((u >> 16) & 1u);
  return (unsigned short)(r >> 16);
}

__device__ inline v8f wmma_bf16(v16bf a, v16bf b, v8f c) {
  // v_wmma_f32_16x16x32_bf16: (neg_a, A, neg_b, B, c_mod, C, reuse_a, reuse_b)
  return __builtin_amdgcn_wmma_f32_16x16x32_bf16(false, a, false, b,
                                                 (short)0, c, false, false);
}

// ---------------------------------------------------------------------------
__global__ void bb_cvt_f32_bf16(const float* __restrict__ src,
                                unsigned short* __restrict__ dst, int n) {
  int i = blockIdx.x * blockDim.x + threadIdx.x;
  if (i < n) dst[i] = f2bf(src[i]);
}

// ---------------------------------------------------------------------------
// Reduce dense [S,S] byte mask to 64x64-block occupancy bytes.
// Grid (NBLK, NBLK); 256 threads each OR 16 bytes of one 64x64 tile.
__global__ void __launch_bounds__(256)
bb_blockmask(const unsigned char* __restrict__ mask,
             unsigned char* __restrict__ blockAny) {
  __shared__ unsigned int any;
  if (threadIdx.x == 0) any = 0u;
  __syncthreads();

  const int row = blockIdx.x * 64 + (threadIdx.x >> 2);
  const int colb = blockIdx.y * 64 + (threadIdx.x & 3) * 16;
  const uint4 m = *(const uint4*)(mask + (size_t)row * S_LEN + colb);
  const unsigned int v = m.x | m.y | m.z | m.w;
  if (v) atomicOr(&any, 1u);
  __syncthreads();

  if (threadIdx.x == 0)
    blockAny[blockIdx.x * NBLK + blockIdx.y] = (unsigned char)(any ? 1 : 0);
}

// ---------------------------------------------------------------------------
// C[M,N] = A[M,K] (bf16, row-major) @ W[N,K]^T (bf16, row-major) + bias[N]
// Block: 256 threads = 8 waves, wave grid 4x2 of 32x32 tiles => 128x64/block.
__global__ void __launch_bounds__(256)
bb_gemm_bf16(const unsigned short* __restrict__ A,
             const unsigned short* __restrict__ W,
             const float* __restrict__ bias,
             unsigned short* __restrict__ outB,
             float* __restrict__ outF,
             int M, int N, int K, int storeF32) {
  const int lane = threadIdx.x & 31;
  const int wave = threadIdx.x >> 5;
  const int col  = lane & 15;
  const int hi   = lane >> 4;
  const int mtile = blockIdx.x * 128 + (wave >> 1) * 32;
  const int ntile = blockIdx.y * 64  + (wave & 1)  * 32;

  v8f acc[2][2];
#pragma unroll
  for (int i = 0; i < 2; ++i)
#pragma unroll
    for (int j = 0; j < 2; ++j)
      acc[i][j] = (v8f){0.f,0.f,0.f,0.f,0.f,0.f,0.f,0.f};

  for (int kb = 0; kb < K; kb += 32) {
    Frag af[2], bf[2];
#pragma unroll
    for (int i = 0; i < 2; ++i) {   // A 16x32: lane=row(M), chunks hi*8, 16+hi*8
      const unsigned short* ap = A + (size_t)(mtile + i * 16 + col) * K + kb + hi * 8;
      af[i].q[0] = *(const uint4*)(ap);
      af[i].q[1] = *(const uint4*)(ap + 16);
    }
#pragma unroll
    for (int j = 0; j < 2; ++j) {   // B 32x16: lane=col(N), chunks hi*8, 16+hi*8
      const unsigned short* bp = W + (size_t)(ntile + j * 16 + col) * K + kb + hi * 8;
      bf[j].q[0] = *(const uint4*)(bp);
      bf[j].q[1] = *(const uint4*)(bp + 16);
    }
#pragma unroll
    for (int i = 0; i < 2; ++i)
#pragma unroll
      for (int j = 0; j < 2; ++j)
        acc[i][j] = wmma_bf16(af[i].v, bf[j].v, acc[i][j]);
  }

#pragma unroll
  for (int i = 0; i < 2; ++i) {
#pragma unroll
    for (int j = 0; j < 2; ++j) {
      const int ccol = ntile + j * 16 + col;
      const float bv = bias[ccol];
#pragma unroll
      for (int r = 0; r < 8; ++r) {
        const int crow = mtile + i * 16 + hi * 8 + r;   // C layout: lane-half picks M+8
        const float val = acc[i][j][r] + bv;
        if (storeF32) outF[(size_t)crow * N + ccol] = val;
        else          outB[(size_t)crow * N + ccol] = f2bf(val);
      }
    }
  }
}

// ---------------------------------------------------------------------------
// Flash attention: grid (S/64, H), block 128 = 4 waves, wave = 16 query rows.
// Skips KV blocks whose 64x64 mask tile is empty (uniform scalar branch).
__global__ void __launch_bounds__(128)
bb_flash_attn(const unsigned short* __restrict__ Q,
              const unsigned short* __restrict__ K,
              const unsigned short* __restrict__ V,
              const unsigned char* __restrict__ mask,
              const unsigned char* __restrict__ blockAny,
              unsigned short* __restrict__ O) {
  __shared__ unsigned short ldsP[4 * 16 * 64];   // per-wave P tiles (bf16)
  __shared__ unsigned short ldsVt[64 * 64];      // V^T tile: [d][kv]

  const int lane  = threadIdx.x & 31;
  const int wave  = threadIdx.x >> 5;
  const int col   = lane & 15;
  const int hi    = lane >> 4;
  const int h     = blockIdx.y;
  const int qblk  = blockIdx.x;
  const int qbase = qblk * 64 + wave * 16;
  const float scale = 0.125f;                    // 1/sqrt(64)

  // Q fragments (held in registers for the whole KV loop)
  Frag qf[2];
  {
    const unsigned short* qp = Q + (size_t)(qbase + col) * C_DIM + h * DH;
#pragma unroll
    for (int s = 0; s < 2; ++s) {
      qf[s].q[0] = *(const uint4*)(qp + s * 32 + hi * 8);
      qf[s].q[1] = *(const uint4*)(qp + s * 32 + 16 + hi * 8);
    }
  }

  v8f oacc[4];
#pragma unroll
  for (int t = 0; t < 4; ++t) oacc[t] = (v8f){0.f,0.f,0.f,0.f,0.f,0.f,0.f,0.f};
  float mrow[8], lrow[8];
#pragma unroll
  for (int r = 0; r < 8; ++r) { mrow[r] = -1e30f; lrow[r] = 0.f; }

  const int rowbase = qbase + hi * 8;            // C-layout row base for this lane
  unsigned short* myP = ldsP + wave * (16 * 64);
  const unsigned char* bmRow = blockAny + qblk * NBLK;

  for (int jb = 0; jb < S_LEN; jb += 64) {
    // BigBird block sparsity: skip empty 64x64 KV tiles (uniform across block)
    if (!bmRow[jb >> 6]) continue;

    __syncthreads();   // previous iteration's LDS reads done before re-staging

    // --- stage V^T into LDS (all 4 waves cooperate; coalesced global reads)
    {
      const int kvrel = threadIdx.x >> 1;        // 0..63
      const int half  = threadIdx.x & 1;         // d half
      const unsigned short* vp =
          V + (size_t)(jb + kvrel) * C_DIM + h * DH + half * 32;
      union { uint4 q[4]; unsigned short u[32]; } vt;
      vt.q[0] = *(const uint4*)(vp + 0);
      vt.q[1] = *(const uint4*)(vp + 8);
      vt.q[2] = *(const uint4*)(vp + 16);
      vt.q[3] = *(const uint4*)(vp + 24);
#pragma unroll
      for (int i = 0; i < 32; ++i)
        ldsVt[(half * 32 + i) * 64 + kvrel] = vt.u[i];
    }

    // --- S tile = Q @ K^T  (8 WMMAs)
    v8f sacc[4];
#pragma unroll
    for (int t = 0; t < 4; ++t) sacc[t] = (v8f){0.f,0.f,0.f,0.f,0.f,0.f,0.f,0.f};
#pragma unroll
    for (int s = 0; s < 2; ++s) {
#pragma unroll
      for (int t = 0; t < 4; ++t) {
        Frag kf;   // B operand: lane = kv column, d chunks hi*8, 16+hi*8
        const unsigned short* kp =
            K + (size_t)(jb + t * 16 + col) * C_DIM + h * DH + s * 32 + hi * 8;
        kf.q[0] = *(const uint4*)(kp);
        kf.q[1] = *(const uint4*)(kp + 16);
        sacc[t] = wmma_bf16(qf[s].v, kf.v, sacc[t]);
      }
    }

    // --- scale + dense byte mask
#pragma unroll
    for (int t = 0; t < 4; ++t) {
      const int kv = jb + t * 16 + col;
#pragma unroll
      for (int r = 0; r < 8; ++r) {
        float v = sacc[t][r] * scale;
        if (!mask[(size_t)(rowbase + r) * S_LEN + kv]) v = -1e30f;
        sacc[t][r] = v;
      }
    }

    // --- online softmax: row max (reduce across the 16 lanes of this half)
    float mnew[8], alpha[8];
#pragma unroll
    for (int r = 0; r < 8; ++r) {
      float v = fmaxf(fmaxf(sacc[0][r], sacc[1][r]),
                      fmaxf(sacc[2][r], sacc[3][r]));
#pragma unroll
      for (int d = 1; d < 16; d <<= 1) v = fmaxf(v, __shfl_xor(v, d, 32));
      const float mn = fmaxf(mrow[r], v);
      alpha[r] = __expf(mrow[r] - mn);
      mnew[r]  = mn;
      mrow[r]  = mn;
    }

    // --- P = exp(S - m): write bf16 tile to per-wave LDS, accumulate row sums
    float psum[8];
#pragma unroll
    for (int r = 0; r < 8; ++r) psum[r] = 0.f;
#pragma unroll
    for (int t = 0; t < 4; ++t) {
#pragma unroll
      for (int r = 0; r < 8; ++r) {
        const float p = __expf(sacc[t][r] - mnew[r]);
        psum[r] += p;
        myP[(hi * 8 + r) * 64 + t * 16 + col] = f2bf(p);
      }
    }
#pragma unroll
    for (int r = 0; r < 8; ++r) {
      float v = psum[r];
#pragma unroll
      for (int d = 1; d < 16; d <<= 1) v += __shfl_xor(v, d, 32);
      lrow[r] = lrow[r] * alpha[r] + v;
    }
#pragma unroll
    for (int t = 0; t < 4; ++t)
#pragma unroll
      for (int r = 0; r < 8; ++r)
        oacc[t][r] *= alpha[r];

    __syncthreads();   // V^T staged + P written before fragment reloads

    // --- reload P as A fragments (LDS, A layout: lane=row, chunks hi*8/16+hi*8)
    Frag pf[2];
#pragma unroll
    for (int s = 0; s < 2; ++s) {
      const unsigned short* pp = myP + (size_t)col * 64 + s * 32 + hi * 8;
      pf[s].q[0] = *(const uint4*)(pp);
      pf[s].q[1] = *(const uint4*)(pp + 16);
    }

    // --- O += P @ V  (8 WMMAs); B operand from transposed V in LDS
#pragma unroll
    for (int s = 0; s < 2; ++s) {
#pragma unroll
      for (int t = 0; t < 4; ++t) {
        Frag vf;
        const unsigned short* vtp =
            ldsVt + (size_t)(t * 16 + col) * 64 + s * 32 + hi * 8;
        vf.q[0] = *(const uint4*)(vtp);
        vf.q[1] = *(const uint4*)(vtp + 16);
        oacc[t] = wmma_bf16(pf[s].v, vf.v, oacc[t]);
      }
    }
  }

  // --- finalize: O / l, store bf16 head-packed [s, h*64 + d]
#pragma unroll
  for (int r = 0; r < 8; ++r) {
    const float inv = 1.0f / fmaxf(lrow[r], 1e-30f);
#pragma unroll
    for (int t = 0; t < 4; ++t) {
      const float val = oacc[t][r] * inv;
      O[(size_t)(rowbase + r) * C_DIM + h * DH + t * 16 + col] = f2bf(val);
    }
  }
}

// ---------------------------------------------------------------------------
extern "C" void kernel_launch(void* const* d_in, const int* in_sizes, int n_in,
                              void* d_out, int out_size, void* d_ws, size_t ws_size,
                              hipStream_t stream) {
  (void)in_sizes; (void)n_in; (void)out_size; (void)ws_size;
  const float*         x    = (const float*)d_in[0];
  const unsigned char* mask = (const unsigned char*)d_in[1];   // jnp bool = 1 byte
  const float* Wq = (const float*)d_in[2];
  const float* bq = (const float*)d_in[3];
  const float* Wk = (const float*)d_in[4];
  const float* bk = (const float*)d_in[5];
  const float* Wv = (const float*)d_in[6];
  const float* bv = (const float*)d_in[7];
  const float* Wo = (const float*)d_in[8];
  const float* bo = (const float*)d_in[9];
  float* out = (float*)d_out;

  // workspace carve-up, 256B aligned
  unsigned char* ws = (unsigned char*)d_ws;
  size_t off = 0;
  auto carve = [&](size_t bytes) -> void* {
    void* p = (void*)(ws + off);
    off += (bytes + 255) & ~(size_t)255;
    return p;
  };
  const size_t xElems = (size_t)S_LEN * C_DIM;     // 2,097,152
  const size_t wElems = (size_t)C_DIM * C_DIM;     //   262,144
  unsigned short* xb  = (unsigned short*)carve(xElems * 2);
  unsigned short* Wqb = (unsigned short*)carve(wElems * 2);
  unsigned short* Wkb = (unsigned short*)carve(wElems * 2);
  unsigned short* Wvb = (unsigned short*)carve(wElems * 2);
  unsigned short* Wob = (unsigned short*)carve(wElems * 2);
  unsigned short* Qb  = (unsigned short*)carve(xElems * 2);
  unsigned short* Kb  = (unsigned short*)carve(xElems * 2);
  unsigned short* Vb  = (unsigned short*)carve(xElems * 2);
  unsigned short* Ao  = (unsigned short*)carve(xElems * 2);
  unsigned char*  bmB = (unsigned char*)carve((size_t)NBLK * NBLK);

  // 1) converts
  {
    const int nx = (int)xElems, nw = (int)wElems;
    bb_cvt_f32_bf16<<<(nx + 255) / 256, 256, 0, stream>>>(x,  xb,  nx);
    bb_cvt_f32_bf16<<<(nw + 255) / 256, 256, 0, stream>>>(Wq, Wqb, nw);
    bb_cvt_f32_bf16<<<(nw + 255) / 256, 256, 0, stream>>>(Wk, Wkb, nw);
    bb_cvt_f32_bf16<<<(nw + 255) / 256, 256, 0, stream>>>(Wv, Wvb, nw);
    bb_cvt_f32_bf16<<<(nw + 255) / 256, 256, 0, stream>>>(Wo, Wob, nw);
  }

  // 2) block-occupancy map of the BigBird mask
  bb_blockmask<<<dim3(NBLK, NBLK), 256, 0, stream>>>(mask, bmB);

  // 3) QKV projections (bf16 out)
  dim3 gg(S_LEN / 128, C_DIM / 64), gb(256);
  bb_gemm_bf16<<<gg, gb, 0, stream>>>(xb, Wqb, bq, Qb, nullptr, S_LEN, C_DIM, C_DIM, 0);
  bb_gemm_bf16<<<gg, gb, 0, stream>>>(xb, Wkb, bk, Kb, nullptr, S_LEN, C_DIM, C_DIM, 0);
  bb_gemm_bf16<<<gg, gb, 0, stream>>>(xb, Wvb, bv, Vb, nullptr, S_LEN, C_DIM, C_DIM, 0);

  // 4) flash attention with block skip
  bb_flash_attn<<<dim3(NBLK, NH), 128, 0, stream>>>(Qb, Kb, Vb, mask, bmB, Ao);

  // 5) output projection (f32 out)
  bb_gemm_bf16<<<gg, gb, 0, stream>>>(Ao, Wob, bo, nullptr, out, S_LEN, C_DIM, C_DIM, 1);
}
